// MambaBlock_541165879762
// MI455X (gfx1250) — compile-verified, matrix operand order checked by reading, not run on hardware
//
#include <hip/hip_runtime.h>
#include <hip/hip_bf16.h>

typedef __attribute__((ext_vector_type(16))) __bf16 v16bf;
typedef __attribute__((ext_vector_type(8)))  float  v8f;

#define D_MODEL 256
#define D_INNER 512
#define N_STATE 16
#define DT_RANK 32
#define NB      4
#define LSEQ    1024
#define NSCAN   4
#define NCHUNK  8
#define LCHUNK  128

static __device__ __forceinline__ unsigned short f32_to_bf16(float f) {
    unsigned int u = __float_as_uint(f);
    unsigned int r = (u + 0x7FFFu + ((u >> 16) & 1u)) >> 16;   // RNE
    return (unsigned short)r;
}

static __device__ __forceinline__ float silu_f(float v) {
    return v * (1.0f / (1.0f + __expf(-v)));
}

// spatial index for sequence position l of scan s (H=W=32)
static __device__ __forceinline__ int sp_of(int s, int l) {
    switch (s) {
        case 0:  return l;
        case 1:  return (LSEQ - 1) - l;
        case 2:  return ((l & 31) << 5) | (l >> 5);
        default: { int m = (LSEQ - 1) - l; return ((m & 31) << 5) | (m >> 5); }
    }
}

// ---------------- conversion kernels (f32 -> bf16, with transpose for weights)
__global__ void k_cvt_x(const float* __restrict__ x, unsigned short* __restrict__ xh, int n) {
    int t = blockIdx.x * blockDim.x + threadIdx.x;
    if (t < n) xh[t] = f32_to_bf16(x[t]);
}
__global__ void k_cvt_wiT(const float* __restrict__ Wi, unsigned short* __restrict__ WiT) {
    int t = blockIdx.x * blockDim.x + threadIdx.x;           // t over 1024*256
    int nn = t >> 8, k = t & 255;
    WiT[t] = f32_to_bf16(Wi[k * (2 * D_INNER) + nn]);
}
__global__ void k_cvt_woT(const float* __restrict__ Wo, unsigned short* __restrict__ WoT) {
    int t = blockIdx.x * blockDim.x + threadIdx.x;           // t over 256*512
    int nn = t >> 9, k = t & 511;
    WoT[t] = f32_to_bf16(Wo[k * D_MODEL + nn]);
}

// ---------------- bf16 WMMA GEMM:  C[M,N] = A[M,K] * B^T   (B stored as [N,K])
// one wave computes a 16x64 strip (4 x 16x16 tiles, A fragment reused)
static __device__ __forceinline__ v16bf load_frag(const unsigned short* __restrict__ row,
                                                  int kk, int hi) {
    v16bf v;
#pragma unroll
    for (int p = 0; p < 8; ++p) {
        int k0 = kk + ((p < 4) ? (2 * p) : (2 * p + 8)) + hi * 8;
        union { unsigned int u; __bf16 h[2]; } cv;
        cv.u = *(const unsigned int*)(row + k0);
        v[2 * p]     = cv.h[0];
        v[2 * p + 1] = cv.h[1];
    }
    return v;
}
static __device__ __forceinline__ v8f wmma_bf16(v16bf a, v16bf b, v8f c) {
    return __builtin_amdgcn_wmma_f32_16x16x32_bf16(false, a, false, b, (short)0, c, false, false);
}

__global__ void k_gemm_bf16(const unsigned short* __restrict__ A,
                            const unsigned short* __restrict__ B,
                            float* __restrict__ C, int M, int N, int K) {
    int wave = blockIdx.x * (blockDim.x >> 5) + (threadIdx.x >> 5);
    int lane = threadIdx.x & 31;
    int nStrips = N >> 6;                       // strips of 64 cols
    int mt = wave / nStrips, ns = wave % nStrips;
    if (mt * 16 >= M) return;
    int hi  = lane >> 4;                        // half-wave
    int l15 = lane & 15;
    const unsigned short* aRow = A + (size_t)(mt * 16 + l15) * K;
    const unsigned short* bR0 = B + (size_t)(ns * 64 +  0 + l15) * K;
    const unsigned short* bR1 = B + (size_t)(ns * 64 + 16 + l15) * K;
    const unsigned short* bR2 = B + (size_t)(ns * 64 + 32 + l15) * K;
    const unsigned short* bR3 = B + (size_t)(ns * 64 + 48 + l15) * K;
    v8f c0 = {}, c1 = {}, c2 = {}, c3 = {};
    for (int kk = 0; kk < K; kk += 32) {
        v16bf a  = load_frag(aRow, kk, hi);
        v16bf b0 = load_frag(bR0, kk, hi);
        v16bf b1 = load_frag(bR1, kk, hi);
        v16bf b2 = load_frag(bR2, kk, hi);
        v16bf b3 = load_frag(bR3, kk, hi);
        c0 = wmma_bf16(a, b0, c0);
        c1 = wmma_bf16(a, b1, c1);
        c2 = wmma_bf16(a, b2, c2);
        c3 = wmma_bf16(a, b3, c3);
    }
    int mrow = mt * 16 + hi * 8;
#pragma unroll
    for (int r = 0; r < 8; ++r) {
        float* cr = C + (size_t)(mrow + r) * N + ns * 64 + l15;
        cr[0]  = c0[r];
        cr[16] = c1[r];
        cr[32] = c2[r];
        cr[48] = c3[r];
    }
}

// ---------------- depthwise causal conv (per scan order) + SiLU
__global__ void k_conv(const float* __restrict__ xz, const float* __restrict__ cw,
                       const float* __restrict__ cb, float* __restrict__ xssm) {
    int t = blockIdx.x * blockDim.x + threadIdx.x;   // over 4*4*1024*512
    int d = t & (D_INNER - 1);
    int l = (t >> 9) & (LSEQ - 1);
    int b = (t >> 19) & 3;
    int s = t >> 21;
    float acc = cb[d];
#pragma unroll
    for (int j = 0; j < 4; ++j) {
        int l2 = l - 3 + j;
        if (l2 >= 0) {
            int sp = sp_of(s, l2);
            acc += cw[d * 4 + j] * xz[((size_t)(b * LSEQ + sp)) * (2 * D_INNER) + d];
        }
    }
    xssm[(size_t)t] = silu_f(acc);
}

// ---------------- x_dbl = x_ssm @ Wx  (K=512 -> 64 outputs)
__global__ void k_xdbl(const float* __restrict__ xssm, const float* __restrict__ Wx,
                       float* __restrict__ xdbl) {
    int t = blockIdx.x * blockDim.x + threadIdx.x;   // over 4*4*1024*64
    int j = t & 63;
    int row = t >> 6;                                // (s*4+b)*1024 + l
    const float* xr = xssm + (size_t)row * D_INNER;
    float acc = 0.f;
    for (int k = 0; k < D_INNER; ++k) acc += xr[k] * Wx[k * 64 + j];
    xdbl[(size_t)t] = acc;
}

// ---------------- SSM scan, chunked. PASS 0: produce per-chunk (P, S).
//                  PASS 1: consume h0, emit y (at spatial index, + Dp*x term).
template <int PASS>
__global__ void k_scan(const float* __restrict__ xssm, const float* __restrict__ xdbl,
                       const float* __restrict__ Wdt, const float* __restrict__ bdt,
                       const float* __restrict__ A_log, const float* __restrict__ Dpv,
                       float* __restrict__ Pb, float* __restrict__ Sb,
                       const float* __restrict__ h0b, float* __restrict__ ybuf) {
    int bi = blockIdx.x;
    int db = bi & 1;
    int c  = (bi >> 1) & 7;
    int b  = (bi >> 4) & 3;
    int s  = bi >> 6;
    int d  = db * 256 + threadIdx.x;

    float wdt[DT_RANK];
#pragma unroll
    for (int r = 0; r < DT_RANK; ++r) wdt[r] = Wdt[r * D_INNER + d];
    float Av[N_STATE];
#pragma unroll
    for (int n = 0; n < N_STATE; ++n) Av[n] = -__expf(A_log[d * N_STATE + n]);
    float bd  = bdt[d];
    float Dpd = Dpv[d];

    int psBase = (((s * 4 + b) * NCHUNK + c) * N_STATE) * D_INNER + d;
    float h[N_STATE], P[N_STATE];
    if (PASS == 0) {
#pragma unroll
        for (int n = 0; n < N_STATE; ++n) { h[n] = 0.f; P[n] = 1.f; }
    } else {
#pragma unroll
        for (int n = 0; n < N_STATE; ++n) h[n] = h0b[psBase + n * D_INNER];
    }

    int l0 = c * LCHUNK;
    for (int l = l0; l < l0 + LCHUNK; ++l) {
        size_t row = (size_t)(s * 4 + b) * LSEQ + l;
        const float* xr = xdbl + row * 64;
        float xv = xssm[row * D_INNER + d];
        // delta = softplus(dt . Wdt[:,d] + bdt[d])
        float acc = bd;
#pragma unroll
        for (int r = 0; r < DT_RANK; ++r) acc += xr[r] * wdt[r];
        float delta = (acc > 20.f) ? acc : __logf(1.f + __expf(acc));
        float dx = delta * xv;
        float y = 0.f;
#pragma unroll
        for (int n = 0; n < N_STATE; ++n) {
            float dA = __expf(delta * Av[n]);
            h[n] = dA * h[n] + dx * xr[32 + n];          // Bp
            if (PASS == 0) P[n] *= dA;
            else           y += h[n] * xr[48 + n];       // Cp
        }
        if (PASS == 1) {
            y += Dpd * xv;
            int sp = sp_of(s, l);
            ybuf[((size_t)(s * 4 + b) * LSEQ + sp) * D_INNER + d] = y;
        }
    }
    if (PASS == 0) {
#pragma unroll
        for (int n = 0; n < N_STATE; ++n) {
            Pb[psBase + n * D_INNER] = P[n];
            Sb[psBase + n * D_INNER] = h[n];
        }
    }
}

// ---------------- serial combine across chunks (tiny): h0[c+1] = P_c*h0[c] + S_c
__global__ void k_scan_combine(const float* __restrict__ Pb, const float* __restrict__ Sb,
                               float* __restrict__ h0b) {
    int t = blockIdx.x * blockDim.x + threadIdx.x;   // over 4*4*512
    int d = t & (D_INNER - 1);
    int b = (t >> 9) & 3;
    int s = t >> 11;
    float h[N_STATE];
#pragma unroll
    for (int n = 0; n < N_STATE; ++n) h[n] = 0.f;
    for (int c = 0; c < NCHUNK; ++c) {
        int base = (((s * 4 + b) * NCHUNK + c) * N_STATE) * D_INNER + d;
#pragma unroll
        for (int n = 0; n < N_STATE; ++n) h0b[base + n * D_INNER] = h[n];
#pragma unroll
        for (int n = 0; n < N_STATE; ++n)
            h[n] = Pb[base + n * D_INNER] * h[n] + Sb[base + n * D_INNER];
    }
}

// ---------------- gate: u = silu(z) * sum_s y_s  -> bf16 for output GEMM
__global__ void k_gate(const float* __restrict__ xz, const float* __restrict__ ybuf,
                       unsigned short* __restrict__ uh) {
    int t = blockIdx.x * blockDim.x + threadIdx.x;   // over 4096*512
    int d = t & (D_INNER - 1);
    int row = t >> 9;                                // b*1024 + sp
    float z = xz[(size_t)row * (2 * D_INNER) + D_INNER + d];
    const size_t stride = (size_t)NB * LSEQ * D_INNER;
    size_t idx = (size_t)row * D_INNER + d;
    float sum = ybuf[idx] + ybuf[idx + stride] + ybuf[idx + 2 * stride] + ybuf[idx + 3 * stride];
    uh[idx] = f32_to_bf16(silu_f(z) * sum);
}

extern "C" void kernel_launch(void* const* d_in, const int* in_sizes, int n_in,
                              void* d_out, int out_size, void* d_ws, size_t ws_size,
                              hipStream_t stream) {
    const float* x      = (const float*)d_in[0];
    const float* Wi     = (const float*)d_in[1];
    const float* conv_w = (const float*)d_in[2];
    const float* conv_b = (const float*)d_in[3];
    const float* Wx     = (const float*)d_in[4];
    const float* Wdt    = (const float*)d_in[5];
    const float* bdt    = (const float*)d_in[6];
    const float* A_log  = (const float*)d_in[7];
    const float* Dp     = (const float*)d_in[8];
    const float* Wo     = (const float*)d_in[9];
    float* out = (float*)d_out;

    const size_t ROWS = (size_t)NB * LSEQ;           // 4096
    char* w = (char*)d_ws;
    float* xz            = (float*)w;  w += ROWS * 2 * D_INNER * 4;          // 16 MB
    unsigned short* Xh   = (unsigned short*)w; w += ROWS * D_MODEL * 2;      // 2 MB
    unsigned short* WiT  = (unsigned short*)w; w += (size_t)2 * D_INNER * D_MODEL * 2;
    unsigned short* WoT  = (unsigned short*)w; w += (size_t)D_MODEL * D_INNER * 2;
    float* xssm          = (float*)w;  w += (size_t)NSCAN * ROWS * D_INNER * 4;   // 32 MB
    float* xdbl          = (float*)w;  w += (size_t)NSCAN * ROWS * 64 * 4;        // 4 MB
    const size_t psElems = (size_t)NSCAN * NB * NCHUNK * N_STATE * D_INNER;
    float* Pb            = (float*)w;  w += psElems * 4;
    float* Sb            = (float*)w;  w += psElems * 4;
    float* h0b           = (float*)w;  w += psElems * 4;
    float* ybuf          = (float*)w;  w += (size_t)NSCAN * ROWS * D_INNER * 4;   // 32 MB
    unsigned short* uh   = (unsigned short*)w; w += ROWS * D_INNER * 2;           // 4 MB
    (void)ws_size; (void)in_sizes; (void)n_in; (void)out_size;

    // 1) bf16 conversions (x) and weight transposes (Wi, Wo)
    k_cvt_x  <<<(ROWS * D_MODEL + 255) / 256, 256, 0, stream>>>(x, Xh, (int)(ROWS * D_MODEL));
    k_cvt_wiT<<<(2 * D_INNER * D_MODEL) / 256, 256, 0, stream>>>(Wi, WiT);
    k_cvt_woT<<<(D_MODEL * D_INNER) / 256, 256, 0, stream>>>(Wo, WoT);

    // 2) xz = x @ Wi   (4096 x 1024, K=256) — shared by all 4 scans
    k_gemm_bf16<<<(ROWS / 16) * (2 * D_INNER / 64) / 4, 128, 0, stream>>>(
        Xh, WiT, xz, (int)ROWS, 2 * D_INNER, D_MODEL);

    // 3) per-scan causal depthwise conv + SiLU (reads xz through the scan perm)
    k_conv<<<(NSCAN * ROWS * D_INNER) / 256, 256, 0, stream>>>(xz, conv_w, conv_b, xssm);

    // 4) x_dbl = x_ssm @ Wx  (dt | B | C)
    k_xdbl<<<(NSCAN * ROWS * 64) / 256, 256, 0, stream>>>(xssm, Wx, xdbl);

    // 5) chunked SSM scan: pass A (chunk products), combine, pass C (emit y)
    k_scan<0><<<NSCAN * NB * NCHUNK * (D_INNER / 256), 256, 0, stream>>>(
        xssm, xdbl, Wdt, bdt, A_log, Dp, Pb, Sb, h0b, ybuf);
    k_scan_combine<<<(NSCAN * NB * D_INNER) / 256, 256, 0, stream>>>(Pb, Sb, h0b);
    k_scan<1><<<NSCAN * NB * NCHUNK * (D_INNER / 256), 256, 0, stream>>>(
        xssm, xdbl, Wdt, bdt, A_log, Dp, Pb, Sb, h0b, ybuf);

    // 6) gate: u = silu(z) * (y0+y1+y2+y3), convert to bf16
    k_gate<<<(ROWS * D_INNER) / 256, 256, 0, stream>>>(xz, ybuf, uh);

    // 7) out = u @ Wo   (4096 x 256, K=512)
    k_gemm_bf16<<<(ROWS / 16) * (D_MODEL / 64) / 4, 128, 0, stream>>>(
        uh, WoT, out, (int)ROWS, D_MODEL, D_INNER);
}
